// MultiDCP_CheMoE_2250562863546
// MI455X (gfx1250) — matrix operand order repaired and checked by reading, not compiled, not run on hardware
//
#include <hip/hip_runtime.h>
#include <hip/hip_bf16.h>

typedef __attribute__((ext_vector_type(16))) _Float16 v16h;
typedef __attribute__((ext_vector_type(8)))  float    v8f;

#define B_    256
#define A_    64
#define G_    978
#define DD    128     // drug / gene embed dim
#define GLOB  306
#define EXPIN 434
#define E_    4
#define H_    128     // expert hidden

#define MT1   16      // P1 M-tiles (256/16)
#define KS1   10      // P1 K-steps (306 -> 320)
#define MT2   62      // P2 M-tiles (978 -> 992)
#define KS2   4       // P2 K-steps (128)
#define NT_   32      // N-tiles total (E*128/16)

// ---- WMMA wave32 per-lane index helpers (ISA 7.12.2, 16-bit A 16x32 / B 32x16) ----
__device__ __forceinline__ int a_k_of(int lane, int j) {
    // lanes 0-15: K in [0..7] U [16..23]; lanes 16-31: K in [8..15] U [24..31]
    int kb = (lane < 16) ? 0 : 8;
    return kb + ((j < 8) ? j : (j + 8));
}
__device__ __forceinline__ int b_k_of(int lane, int j) {
    // lanes 0-15: K in [0..7] U [16..23]; lanes 16-31: K in [8..15] U [24..31]
    return ((lane < 16) ? 0 : 8) + ((j < 8) ? j : (j + 8));
}

// ============================================================================
// Kernel 1: fused small encoders (drug sum + cell MLP + dose MLP) -> g_feat.
// ============================================================================
__global__ void encoders_kernel(const float* __restrict__ drug_atom,
                                const float* __restrict__ cell_gex,
                                const float* __restrict__ dose_in,
                                const float* __restrict__ cW1, const float* __restrict__ cb1,
                                const float* __restrict__ cW2, const float* __restrict__ cb2,
                                const float* __restrict__ cW3, const float* __restrict__ cb3,
                                const float* __restrict__ dW1, const float* __restrict__ db1,
                                const float* __restrict__ dW2, const float* __restrict__ db2,
                                float* __restrict__ g_feat,
                                float* __restrict__ out_cell)
{
    __shared__ float s_in[978];
    __shared__ float s_h1[200];
    __shared__ float s_h2[100];
    __shared__ float s_d[12];
    __shared__ float s_dh[64];
    const int b = blockIdx.x, tid = threadIdx.x;

    for (int k = tid; k < 978; k += 256) s_in[k] = cell_gex[b * 978 + k];
    if (tid < 12) s_d[tid] = dose_in[b * 12 + tid];

    if (tid < 128) {  // drug atom-sum
        float s = 0.f;
        const float* p = drug_atom + (size_t)b * A_ * DD + tid;
        for (int a = 0; a < A_; ++a) s += p[a * DD];
        g_feat[b * GLOB + tid] = s;
    }
    __syncthreads();

    if (tid < 200) {  // cell L1
        float acc = cb1[tid];
        for (int k = 0; k < 978; ++k) acc = fmaf(s_in[k], cW1[k * 200 + tid], acc);
        s_h1[tid] = fmaxf(acc, 0.f);
    }
    if (tid < 64) {   // dose L1
        float acc = db1[tid];
        for (int k = 0; k < 12; ++k) acc = fmaf(s_d[k], dW1[k * 64 + tid], acc);
        s_dh[tid] = fmaxf(acc, 0.f);
    }
    __syncthreads();

    if (tid < 100) {  // cell L2
        float acc = cb2[tid];
        for (int k = 0; k < 200; ++k) acc = fmaf(s_h1[k], cW2[k * 100 + tid], acc);
        s_h2[tid] = fmaxf(acc, 0.f);
    }
    if (tid < 128) {  // dose L2
        float acc = db2[tid];
        for (int k = 0; k < 64; ++k) acc = fmaf(s_dh[k], dW2[k * 128 + tid], acc);
        g_feat[b * GLOB + 178 + tid] = fmaxf(acc, 0.f);
    }
    __syncthreads();

    if (tid < 50) {   // cell L3
        float acc = cb3[tid];
        for (int k = 0; k < 100; ++k) acc = fmaf(s_h2[k], cW3[k * 50 + tid], acc);
        float v = fmaxf(acc, 0.f);
        g_feat[b * GLOB + 128 + tid] = v;
        out_cell[b * 50 + tid] = v;
    }
}

// ============================================================================
// Kernel 2: gating network + top-2 softmax + gate_bias[b]=sum_e gate_e*b2[e].
// ============================================================================
__global__ void gating_kernel(const float* __restrict__ g_feat,
                              const float* __restrict__ gW1, const float* __restrict__ gb1,
                              const float* __restrict__ gW2, const float* __restrict__ gb2,
                              const float* __restrict__ exp_b2,
                              float* __restrict__ gates, float* __restrict__ gate_bias)
{
    __shared__ float s_g[GLOB];
    __shared__ float s_h[128];
    __shared__ float s_l[E_];
    const int b = blockIdx.x, tid = threadIdx.x;

    for (int k = tid; k < GLOB; k += 128) s_g[k] = g_feat[b * GLOB + k];
    __syncthreads();

    float acc = gb1[tid];
    for (int k = 0; k < GLOB; ++k) acc = fmaf(s_g[k], gW1[k * 128 + tid], acc);
    s_h[tid] = fmaxf(acc, 0.f);
    __syncthreads();

    if (tid < E_) {
        float l = gb2[tid];
        for (int k = 0; k < 128; ++k) l = fmaf(s_h[k], gW2[k * E_ + tid], l);
        s_l[tid] = l;
    }
    __syncthreads();

    if (tid == 0) {
        int i0 = 0;
        for (int e = 1; e < E_; ++e) if (s_l[e] > s_l[i0]) i0 = e;
        int i1 = -1;
        for (int e = 0; e < E_; ++e)
            if (e != i0 && (i1 < 0 || s_l[e] > s_l[i1])) i1 = e;
        float m  = fmaxf(s_l[i0], s_l[i1]);
        float e0 = __expf(s_l[i0] - m), e1 = __expf(s_l[i1] - m);
        float inv = 1.f / (e0 + e1);
        float o[E_] = {0.f, 0.f, 0.f, 0.f};
        o[i0] = e0 * inv; o[i1] = e1 * inv;
        for (int e = 0; e < E_; ++e) gates[b * E_ + e] = o[e];
        gate_bias[b] = o[i0] * exp_b2[i0] + o[i1] * exp_b2[i1];
    }
}

// ============================================================================
// Packing kernels: convert f32 operands to f16 pre-swizzled into the wave32
// WMMA register layout (one thread = one lane's 16 halfs = 32B contiguous).
// All K-padding zeros are baked in here, so WMMA kernels have zero guards.
// ============================================================================

// A1h[((mt*KS1+ks)*32 + lane)*16 + j] = f16(g_feat[mt*16+(lane&15), k]), k<306
__global__ void pack_A1_kernel(const float* __restrict__ g_feat,
                               _Float16* __restrict__ A1h)
{
    const int t = blockIdx.x * blockDim.x + threadIdx.x;   // MT1*KS1*32 = 5120
    const int lane = t & 31;
    const int ks   = (t >> 5) % KS1;
    const int mt   = (t >> 5) / KS1;
    const int row  = mt * 16 + (lane & 15);
    _Float16* dst = A1h + (size_t)t * 16;
#pragma unroll
    for (int j = 0; j < 16; ++j) {
        int k = ks * 32 + a_k_of(lane, j);
        dst[j] = (_Float16)((k < GLOB) ? g_feat[row * GLOB + k] : 0.f);
    }
}

// A2h[((mt*KS2+ks)*32 + lane)*16 + j] = f16(gene_emb[mt*16+(lane&15), k]), g<978
__global__ void pack_A2_kernel(const float* __restrict__ gene_emb,
                               _Float16* __restrict__ A2h)
{
    const int t = blockIdx.x * blockDim.x + threadIdx.x;   // MT2*KS2*32 = 7936
    const int lane = t & 31;
    const int ks   = (t >> 5) % KS2;
    const int mt   = (t >> 5) / KS2;
    const int g    = mt * 16 + (lane & 15);
    _Float16* dst = A2h + (size_t)t * 16;
#pragma unroll
    for (int j = 0; j < 16; ++j) {
        int k = ks * 32 + a_k_of(lane, j);
        dst[j] = (_Float16)((g < G_) ? gene_emb[g * DD + k] : 0.f);
    }
}

// B1h: expW1 rows [0,306) (head), padded to 320.  tile id = (e*8+ntn)
// B1h[((tile*KS1+ks)*32 + lane)*16 + j] = f16(W[e][k][ntn*16 + (lane&15)])
__global__ void pack_B1_kernel(const float* __restrict__ expW1,
                               _Float16* __restrict__ B1h)
{
    const int t = blockIdx.x * blockDim.x + threadIdx.x;   // 4*8*KS1*32 = 10240
    const int lane = t & 31;
    const int ks   = (t >> 5) % KS1;
    const int tile = (t >> 5) / KS1;                       // e*8 + ntn
    const int e    = tile >> 3;
    const int col  = (tile & 7) * 16 + (lane & 15);
    const float* W = expW1 + (size_t)e * EXPIN * H_;
    _Float16* dst = B1h + (size_t)t * 16;
#pragma unroll
    for (int j = 0; j < 16; ++j) {
        int k = ks * 32 + b_k_of(lane, j);
        dst[j] = (_Float16)((k < GLOB) ? W[(size_t)k * H_ + col] : 0.f);
    }
}

// B2h: expW1 rows [306,434) (tail), K=128 exact.
__global__ void pack_B2_kernel(const float* __restrict__ expW1,
                               _Float16* __restrict__ B2h)
{
    const int t = blockIdx.x * blockDim.x + threadIdx.x;   // 4*8*KS2*32 = 4096
    const int lane = t & 31;
    const int ks   = (t >> 5) % KS2;
    const int tile = (t >> 5) / KS2;
    const int e    = tile >> 3;
    const int col  = (tile & 7) * 16 + (lane & 15);
    const float* W = expW1 + (size_t)e * EXPIN * H_ + (size_t)GLOB * H_;
    _Float16* dst = B2h + (size_t)t * 16;
#pragma unroll
    for (int j = 0; j < 16; ++j) {
        int k = ks * 32 + b_k_of(lane, j);
        dst[j] = (_Float16)W[(size_t)k * H_ + col];
    }
}

// ============================================================================
// Kernel 3: P1b = g_feat @ expW1_head + b1.  Pure WMMA over packed f16.
// One wave per 16x16 tile; inner loop = 2 packed loads + 1 WMMA.
// ============================================================================
__global__ void p1_wmma_kernel(const _Float16* __restrict__ A1h,
                               const _Float16* __restrict__ B1h,
                               const float* __restrict__ expb1,
                               float* __restrict__ P1b)
{
    const int lane = threadIdx.x & 31;
    const int w    = blockIdx.x * (blockDim.x >> 5) + (threadIdx.x >> 5);
    const int mt = w >> 5;                 // 0..15
    const int nt = w & 31;                 // 0..31  (== e*8 + ntn)
    const int e  = nt >> 3;
    const int nb = (nt & 7) * 16;
    const int n  = lane & 15;

    const v16h* Ap = (const v16h*)(A1h + ((size_t)(mt * KS1) * 32 + lane) * 16);
    const v16h* Bp = (const v16h*)(B1h + ((size_t)(nt * KS1) * 32 + lane) * 16);

    v8f c = {};
#pragma unroll
    for (int ks = 0; ks < KS1; ++ks) {
        v16h a = Ap[ks * 32];
        v16h b = Bp[ks * 32];
        c = __builtin_amdgcn_wmma_f32_16x16x32_f16(false, a, false, b,
                                                   (short)0, c, false, false);
    }
    const float b1v = expb1[e * H_ + nb + n];
#pragma unroll
    for (int r = 0; r < 8; ++r) {
        int mm = r + ((lane < 16) ? 0 : 8);
        P1b[(size_t)(mt * 16 + mm) * (E_ * H_) + e * H_ + nb + n] = c[r] + b1v;
    }
}

// ============================================================================
// Kernel 4: P2[e,g,:] = gene_emb @ expW1_tail.  Pure WMMA over packed f16.
// ============================================================================
__global__ void p2_wmma_kernel(const _Float16* __restrict__ A2h,
                               const _Float16* __restrict__ B2h,
                               float* __restrict__ P2)
{
    const int lane = threadIdx.x & 31;
    const int w    = blockIdx.x * (blockDim.x >> 5) + (threadIdx.x >> 5);
    const int mt = w >> 5;                 // 0..61
    const int nt = w & 31;
    const int e  = nt >> 3;
    const int nb = (nt & 7) * 16;
    const int gbase = mt * 16;

    const v16h* Ap = (const v16h*)(A2h + ((size_t)(mt * KS2) * 32 + lane) * 16);
    const v16h* Bp = (const v16h*)(B2h + ((size_t)(nt * KS2) * 32 + lane) * 16);

    v8f c = {};
#pragma unroll
    for (int ks = 0; ks < KS2; ++ks) {
        v16h a = Ap[ks * 32];
        v16h b = Bp[ks * 32];
        c = __builtin_amdgcn_wmma_f32_16x16x32_f16(false, a, false, b,
                                                   (short)0, c, false, false);
    }
#pragma unroll
    for (int r = 0; r < 8; ++r) {
        int gg = gbase + r + ((lane < 16) ? 0 : 8);
        if (gg < G_) P2[((size_t)e * G_ + gg) * H_ + nb + (lane & 15)] = c[r];
    }
}

// ============================================================================
// Kernel 5: fused MoE combine.
//  pred[b,g] = gate_bias[b] + sum_e gates[b,e] *
//              sum_k relu(P1b[b,e,k] + P2[e,g,k]) * expW2[e,k]
// One wave per gene; lanes over k (float4); top-2 gate skip (wave-uniform).
// ============================================================================
__global__ void moe_final_kernel(const float* __restrict__ P1b,
                                 const float* __restrict__ P2,
                                 const float* __restrict__ expW2,
                                 const float* __restrict__ gates,
                                 const float* __restrict__ gate_bias,
                                 float* __restrict__ out_pred)
{
    __shared__ float s_p1[E_ * H_];
    __shared__ float s_w2[E_ * H_];
    __shared__ float s_g[E_];
    __shared__ float s_bias;
    const int b = blockIdx.y;
    const int tid = threadIdx.x;

    for (int i = tid; i < E_ * H_; i += 256) {
        s_p1[i] = P1b[(size_t)b * E_ * H_ + i];
        s_w2[i] = expW2[i];
    }
    if (tid < E_) s_g[tid] = gates[b * E_ + tid];
    if (tid == 0) s_bias = gate_bias[b];
    __syncthreads();

    const int wave = tid >> 5, lane = tid & 31;
    const int g = blockIdx.x * 8 + wave;
    if (g >= G_) return;

    float acc = 0.f;
#pragma unroll
    for (int e = 0; e < E_; ++e) {
        const float gt = s_g[e];                 // uniform across wave
        if (gt == 0.f) continue;                 // top-2 sparsity
        const float4 p2 =
            *(const float4*)&P2[((size_t)e * G_ + g) * H_ + lane * 4];
        const int base = e * H_ + lane * 4;
        float s = 0.f, t;
        t = s_p1[base + 0] + p2.x; s = fmaf(fmaxf(t, 0.f), s_w2[base + 0], s);
        t = s_p1[base + 1] + p2.y; s = fmaf(fmaxf(t, 0.f), s_w2[base + 1], s);
        t = s_p1[base + 2] + p2.z; s = fmaf(fmaxf(t, 0.f), s_w2[base + 2], s);
        t = s_p1[base + 3] + p2.w; s = fmaf(fmaxf(t, 0.f), s_w2[base + 3], s);
        acc = fmaf(gt, s, acc);
    }
    for (int off = 16; off > 0; off >>= 1) acc += __shfl_down(acc, off, 32);
    if (lane == 0) out_pred[(size_t)b * G_ + g] = acc + s_bias;
}

// ============================================================================
extern "C" void kernel_launch(void* const* d_in, const int* in_sizes, int n_in,
                              void* d_out, int out_size, void* d_ws, size_t ws_size,
                              hipStream_t stream)
{
    (void)in_sizes; (void)n_in; (void)out_size; (void)ws_size;
    const float* drug  = (const float*)d_in[0];
    const float* cellg = (const float*)d_in[1];
    const float* dose  = (const float*)d_in[2];
    const float* cW1 = (const float*)d_in[3],  *cb1 = (const float*)d_in[4];
    const float* cW2 = (const float*)d_in[5],  *cb2 = (const float*)d_in[6];
    const float* cW3 = (const float*)d_in[7],  *cb3 = (const float*)d_in[8];
    const float* dW1 = (const float*)d_in[9],  *db1 = (const float*)d_in[10];
    const float* dW2 = (const float*)d_in[11], *db2 = (const float*)d_in[12];
    const float* gene = (const float*)d_in[13];
    const float* gW1 = (const float*)d_in[14], *gb1 = (const float*)d_in[15];
    const float* gW2 = (const float*)d_in[16], *gb2 = (const float*)d_in[17];
    const float* eW1 = (const float*)d_in[18], *eb1 = (const float*)d_in[19];
    const float* eW2 = (const float*)d_in[20], *eb2 = (const float*)d_in[21];

    // ---- workspace layout (f32 region, then 32B-aligned f16 region) ----
    float* ws     = (float*)d_ws;
    float* g_feat = ws;                          // 256*306   = 78336
    float* gates  = g_feat + 256 * GLOB;         // 1024
    float* gbias  = gates + 256 * E_;            // 256
    float* P1b    = gbias + 256;                 // 256*512   = 131072
    float* P2     = P1b + 256 * E_ * H_;         // 4*978*128 = 500736
    _Float16* A1h = (_Float16*)(P2 + (size_t)E_ * G_ * H_);  // 5120*16  halfs
    _Float16* A2h = A1h + (size_t)MT1 * KS1 * 32 * 16;       // 7936*16
    _Float16* B1h = A2h + (size_t)MT2 * KS2 * 32 * 16;       // 10240*16
    _Float16* B2h = B1h + (size_t)NT_ * KS1 * 32 * 16;       // 4096*16

    float* out_pred = (float*)d_out;
    float* out_cell = out_pred + (size_t)B_ * G_;

    // stage 1: encoders -> g_feat (and cell output)
    encoders_kernel<<<B_, 256, 0, stream>>>(drug, cellg, dose,
                                            cW1, cb1, cW2, cb2, cW3, cb3,
                                            dW1, db1, dW2, db2, g_feat, out_cell);
    // stage 2 (independent of g_feat): pack static operands to f16 WMMA layout
    pack_A2_kernel<<<(MT2 * KS2 * 32) / 256, 256, 0, stream>>>(gene, A2h);
    pack_B1_kernel<<<(NT_ * KS1 * 32) / 256, 256, 0, stream>>>(eW1, B1h);
    pack_B2_kernel<<<(NT_ * KS2 * 32) / 256, 256, 0, stream>>>(eW1, B2h);
    // stage 3: gating + pack dynamic A1 (both need g_feat)
    gating_kernel<<<B_, 128, 0, stream>>>(g_feat, gW1, gb1, gW2, gb2, eb2,
                                          gates, gbias);
    pack_A1_kernel<<<(MT1 * KS1 * 32) / 256, 256, 0, stream>>>(g_feat, A1h);
    // stage 4: WMMA GEMMs (16*32=512 waves; 62*32=1984 waves; 4 waves/block)
    p1_wmma_kernel<<<128, 128, 0, stream>>>(A1h, B1h, eb1, P1b);
    p2_wmma_kernel<<<496, 128, 0, stream>>>(A2h, B2h, P2);
    // stage 5: fused relu-dot-gate combine
    dim3 grid((G_ + 7) / 8, B_);
    moe_final_kernel<<<grid, 256, 0, stream>>>(P1b, P2, eW2, gates, gbias, out_pred);
}
